// GCT_1425929142813
// MI455X (gfx1250) — matrix operand order repaired and checked by reading, not compiled, not run on hardware
//
#include <hip/hip_runtime.h>
#include <hip/hip_bf16.h>
#include <stdint.h>

// ---------------- static problem config ----------------
#define NN   32          // time/batch
#define SS   32          // spatial side
#define CC   64          // hidden
#define HH   4           // heads
#define DD   16          // head dim
#define TK   3           // time kernel
#define KK   3           // spatial kernel
#define MM   27          // TK*KK*KK
#define NLAY 2
#define FF   3
#define PP   (NN*SS*SS)  // 32768 positions
#define SSTRIDE 28       // padded score stride (27 -> 28)

typedef __attribute__((ext_vector_type(16))) _Float16 v16h;
typedef __attribute__((ext_vector_type(8)))  _Float16 v8h;
typedef __attribute__((ext_vector_type(8)))  float    v8f;

// gfx1250 async global->LDS path (ASYNCcnt tracked), guarded so the kernel
// still builds on toolchains without the builtins.
#if defined(__has_builtin)
# if __has_builtin(__builtin_amdgcn_global_load_async_to_lds_b128) && \
     __has_builtin(__builtin_amdgcn_s_wait_asynccnt)
#  define USE_ASYNC_LDS 1
# endif
#endif

#ifdef USE_ASYNC_LDS
// exact parameter types per hipcc diagnostic: AS(1)/AS(3) pointers to gcc-vector int4
typedef int vi4 __attribute__((__vector_size__(16)));
typedef __attribute__((address_space(1))) vi4* gvi4p;
typedef __attribute__((address_space(3))) vi4* lvi4p;
#endif

// ---------------- BatchNorm statistics ----------------
__global__ __launch_bounds__(256) void bn_stats_k(const float* __restrict__ x,
                                                  float* __restrict__ stats) {
    __shared__ float red[256];
    float sum[FF] = {0.f, 0.f, 0.f};
    float sq[FF]  = {0.f, 0.f, 0.f};
    for (int p = threadIdx.x; p < PP; p += 256) {
        #pragma unroll
        for (int f = 0; f < FF; ++f) {
            float v = x[p * FF + f];
            sum[f] += v; sq[f] += v * v;
        }
    }
    #pragma unroll
    for (int f = 0; f < FF; ++f) {
        red[threadIdx.x] = sum[f];
        __syncthreads();
        for (int s = 128; s > 0; s >>= 1) {
            if (threadIdx.x < s) red[threadIdx.x] += red[threadIdx.x + s];
            __syncthreads();
        }
        float tot = red[0];
        __syncthreads();
        red[threadIdx.x] = sq[f];
        __syncthreads();
        for (int s = 128; s > 0; s >>= 1) {
            if (threadIdx.x < s) red[threadIdx.x] += red[threadIdx.x + s];
            __syncthreads();
        }
        float tot2 = red[0];
        __syncthreads();
        if (threadIdx.x == 0) {
            float mean = tot / (float)PP;
            float var  = tot2 / (float)PP - mean * mean;
            stats[f]     = mean;
            stats[4 + f] = rsqrtf(var + 1e-5f);
        }
    }
}

// ---------------- input projection: h = BN(x) @ W_in + b_in ----------------
__global__ __launch_bounds__(256) void input_proj_k(const float* __restrict__ x,
                                                    const float* __restrict__ W_in,
                                                    const float* __restrict__ b_in,
                                                    const float* __restrict__ stats,
                                                    float* __restrict__ h,
                                                    _Float16* __restrict__ h16) {
    int tid = blockIdx.x * 256 + threadIdx.x;   // PP*CC threads
    int p = tid >> 6, c = tid & 63;
    float acc = b_in[c];
    #pragma unroll
    for (int f = 0; f < FF; ++f) {
        float xn = (x[p * FF + f] - stats[f]) * stats[4 + f];
        acc += xn * W_in[f * CC + c];
    }
    h[tid]   = acc;
    h16[tid] = (_Float16)acc;
}

// ---------------- weight conversion: f32 [C][C] -> f16 transposed [d][c] ----
// slots per layer: g=0 -> Wq ; g=1..27 -> Wk[m] ; g=28..54 -> Wv[m]
__global__ __launch_bounds__(256) void wconv_k(const float* __restrict__ Wq,
                                               const float* __restrict__ Wk,
                                               const float* __restrict__ Wv,
                                               _Float16* __restrict__ w16) {
    int tid = blockIdx.x * 256 + threadIdx.x;   // NLAY*55*4096 threads
    if (tid >= NLAY * 55 * 4096) return;
    int l = tid / (55 * 4096);
    int r = tid - l * (55 * 4096);
    int g = r >> 12;
    int e = r & 4095;
    int d = e >> 6, c = e & 63;
    float v;
    if (g == 0)       v = Wq[(size_t)l * 4096 + c * 64 + d];
    else if (g < 28)  v = Wk[((size_t)l * MM + (g - 1)) * 4096 + c * 64 + d];
    else              v = Wv[((size_t)l * MM + (g - 28)) * 4096 + c * 64 + d];
    w16[tid] = (_Float16)v;
}

// ---------------- WMMA GEMM: Y_g = A[32768x64] @ W_g[64x64] + b_g ----------
// A: f16 row-major.  Wt: f16 transposed [g][d][c].  bias: f32 [g][64].
// 128 threads = 4 waves; each wave: 32 rows x 64 cols (8 tiles, 2 k-steps).
__global__ __launch_bounds__(128) void gemm64_k(const _Float16* __restrict__ A,
                                                const _Float16* __restrict__ Wt,
                                                const float* __restrict__ bias,
                                                float* __restrict__ outF32,
                                                _Float16* __restrict__ outF16,
                                                int f16out) {
    __shared__ _Float16 Bs[CC * CC];            // 8 KB weight tile
    const int g = blockIdx.y;

    // warm L2/L0 with this block's A rows while the weight tile is staged
    __builtin_prefetch(A + (size_t)(blockIdx.x * 128 + threadIdx.x) * CC, 0, 3);

    {   // stage weight tile to LDS: 4096 halves = 512 x b128
        const uint4* src = (const uint4*)(Wt + (size_t)g * 4096);
        uint4* dst = (uint4*)Bs;
#ifdef USE_ASYNC_LDS
        #pragma unroll
        for (int i = 0; i < 4; ++i) {
            int idx = threadIdx.x + i * 128;
            __builtin_amdgcn_global_load_async_to_lds_b128(
                (gvi4p)(src + idx), (lvi4p)(dst + idx), 0, 0);
        }
        __builtin_amdgcn_s_wait_asynccnt(0);
#else
        #pragma unroll
        for (int i = 0; i < 4; ++i) dst[threadIdx.x + i * 128] = src[threadIdx.x + i * 128];
#endif
    }
    __syncthreads();

    const int wave = threadIdx.x >> 5;
    const int lane = threadIdx.x & 31;
    const int lrow = lane & 15;
    const int lhi  = lane >> 4;                 // half-wave select
    const int rowBase = blockIdx.x * 128 + wave * 32;

    float bcol[4];
    #pragma unroll
    for (int nt = 0; nt < 4; ++nt) bcol[nt] = bias[g * 64 + nt * 16 + lrow];

    #pragma unroll
    for (int rt = 0; rt < 2; ++rt) {
        const int trow = rowBase + rt * 16;
        const _Float16* ap = A + (size_t)(trow + lrow) * CC;
        // A layout (16-bit 16x32): lanes0-15 K{0-7,16-23}; lanes16-31 K{8-15,24-31}
        v8h a00 = *(const v8h*)(ap + 0  + lhi * 8);
        v8h a01 = *(const v8h*)(ap + 16 + lhi * 8);
        v8h a10 = *(const v8h*)(ap + 32 + lhi * 8);
        v8h a11 = *(const v8h*)(ap + 48 + lhi * 8);
        v16h A0, A1;
        #pragma unroll
        for (int i = 0; i < 8; ++i) {
            A0[i] = a00[i]; A0[8 + i] = a01[i];
            A1[i] = a10[i]; A1[8 + i] = a11[i];
        }
        #pragma unroll
        for (int nt = 0; nt < 4; ++nt) {
            const int col = nt * 16 + lrow;
            // B layout (16-bit 32x16): lane holds column `col`, K = k0 + {lhi*8..+8, 16+lhi*8..+8}
            const _Float16* bp0 = Bs + col * CC + lhi * 8;         // k0 = 0
            const _Float16* bp1 = Bs + col * CC + 32 + lhi * 8;    // k0 = 32
            v8h b00 = *(const v8h*)(bp0);
            v8h b01 = *(const v8h*)(bp0 + 16);
            v8h b10 = *(const v8h*)(bp1);
            v8h b11 = *(const v8h*)(bp1 + 16);
            v16h B0, B1;
            #pragma unroll
            for (int i = 0; i < 8; ++i) {
                B0[i] = b00[i]; B0[8 + i] = b01[i];
                B1[i] = b10[i]; B1[8 + i] = b11[i];
            }
            v8f acc;
            #pragma unroll
            for (int i = 0; i < 8; ++i) acc[i] = bcol[nt];          // bias epilogue folded in
            acc = __builtin_amdgcn_wmma_f32_16x16x32_f16(false, A0, false, B0,
                                                         (short)0, acc, false, false);
            acc = __builtin_amdgcn_wmma_f32_16x16x32_f16(false, A1, false, B1,
                                                         (short)0, acc, false, false);
            // D layout: VGPR i -> row trow + i + lhi*8, col = nt*16 + lrow
            if (f16out) {
                _Float16* op = outF16 + (size_t)g * PP * CC;
                #pragma unroll
                for (int i = 0; i < 8; ++i)
                    op[(size_t)(trow + i + lhi * 8) * CC + col] = (_Float16)acc[i];
            } else {
                #pragma unroll
                for (int i = 0; i < 8; ++i)
                    outF32[(size_t)(trow + i + lhi * 8) * CC + col] = acc[i];
            }
        }
    }
}

// ---------------- scores for one time slice t ----------------
__global__ __launch_bounds__(256) void score_k(const float* __restrict__ Q,
                                               const _Float16* __restrict__ Kt,
                                               const float* __restrict__ bk_lt,
                                               float* __restrict__ S, int t) {
    int tid = blockIdx.x * 256 + threadIdx.x;   // PP*HH threads
    int p = tid >> 2, h = tid & 3;
    int n = p >> 10, a = (p >> 5) & 31, b = p & 31;
    float q[DD];
    const float* qp = Q + (size_t)p * CC + h * DD;
    #pragma unroll
    for (int d = 0; d < DD; ++d) q[d] = qp[d];
    int na = n + t - 1;
    bool tvalid = (na >= 0) && (na < NN);
    float* Sp = S + (size_t)tid * SSTRIDE + t * 9;
    #pragma unroll
    for (int j = 0; j < 9; ++j) {
        const int i = j / 3, jj = j % 3;
        int aa = a + i - 1, bb = b + jj - 1;
        float sc;
        if (aa < 0 || aa >= SS || bb < 0 || bb >= SS) {
            sc = -1e30f;                                    // spatial mask
        } else if (!tvalid) {
            const float* kb = bk_lt + j * CC + h * DD;      // zero-padded time -> bias only
            sc = 0.f;
            #pragma unroll
            for (int d = 0; d < DD; ++d) sc += q[d] * kb[d];
        } else {
            size_t pm = (size_t)(na * 1024 + aa * 32 + bb);
            const v8h* kp = (const v8h*)(Kt + ((size_t)j * PP + pm) * CC + h * DD);
            v8h k0 = kp[0], k1 = kp[1];
            sc = 0.f;
            #pragma unroll
            for (int d = 0; d < 8; ++d) sc += q[d] * (float)k0[d];
            #pragma unroll
            for (int d = 0; d < 8; ++d) sc += q[8 + d] * (float)k1[d];
        }
        Sp[j] = sc;
    }
}

// ---------------- softmax over the 27 offsets ----------------
__global__ __launch_bounds__(256) void softmax_k(float* __restrict__ S) {
    int tid = blockIdx.x * 256 + threadIdx.x;   // PP*HH threads
    float* Sp = S + (size_t)tid * SSTRIDE;
    float mx = -1e30f;
    float v[MM];
    #pragma unroll
    for (int m = 0; m < MM; ++m) { v[m] = Sp[m]; mx = fmaxf(mx, v[m]); }
    float sum = 0.f;
    #pragma unroll
    for (int m = 0; m < MM; ++m) { v[m] = __expf(v[m] - mx); sum += v[m]; }
    float inv = 1.f / sum;
    #pragma unroll
    for (int m = 0; m < MM; ++m) Sp[m] = v[m] * inv;
}

// ---------------- weighted V accumulation for one time slice ----------------
__global__ __launch_bounds__(256) void accum_k(const float* __restrict__ S,
                                               const _Float16* __restrict__ Vt,
                                               const float* __restrict__ bv_lt,
                                               float* __restrict__ O,
                                               const float* __restrict__ hcur,
                                               float* __restrict__ hnext,
                                               _Float16* __restrict__ h16next,
                                               int t, int init, int finalize) {
    int tid = blockIdx.x * 256 + threadIdx.x;   // PP*HH threads
    int p = tid >> 2, h = tid & 3;
    int n = p >> 10, a = (p >> 5) & 31, b = p & 31;
    int na = n + t - 1;
    bool tvalid = (na >= 0) && (na < NN);
    size_t obase = (size_t)p * CC + h * DD;
    float acc[DD];
    if (init) {
        #pragma unroll
        for (int d = 0; d < DD; ++d) acc[d] = 0.f;
    } else {
        #pragma unroll
        for (int d = 0; d < DD; ++d) acc[d] = O[obase + d];
    }
    const float* Wp = S + (size_t)tid * SSTRIDE + t * 9;
    #pragma unroll
    for (int j = 0; j < 9; ++j) {
        const int i = j / 3, jj = j % 3;
        int aa = a + i - 1, bb = b + jj - 1;
        if (aa < 0 || aa >= SS || bb < 0 || bb >= SS) continue;  // weight == 0
        float w = Wp[j];
        if (!tvalid) {
            const float* vb = bv_lt + j * CC + h * DD;
            #pragma unroll
            for (int d = 0; d < DD; ++d) acc[d] += w * vb[d];
        } else {
            size_t pm = (size_t)(na * 1024 + aa * 32 + bb);
            const v8h* vp = (const v8h*)(Vt + ((size_t)j * PP + pm) * CC + h * DD);
            v8h v0 = vp[0], v1 = vp[1];
            #pragma unroll
            for (int d = 0; d < 8; ++d) acc[d] += w * (float)v0[d];
            #pragma unroll
            for (int d = 0; d < 8; ++d) acc[8 + d] += w * (float)v1[d];
        }
    }
    if (finalize) {
        #pragma unroll
        for (int d = 0; d < DD; ++d) {
            float r = acc[d] + hcur[obase + d];                 // residual
            hnext[obase + d]   = r;
            h16next[obase + d] = (_Float16)r;
        }
    } else {
        #pragma unroll
        for (int d = 0; d < DD; ++d) O[obase + d] = acc[d];
    }
}

// ---------------- output projection ----------------
__global__ __launch_bounds__(256) void output_proj_k(const float* __restrict__ h,
                                                     const float* __restrict__ W_out,
                                                     const float* __restrict__ b_out,
                                                     float* __restrict__ out) {
    int tid = blockIdx.x * 256 + threadIdx.x;   // PP*FF threads
    int p = tid / FF, f = tid - p * FF;
    const float* hp = h + (size_t)p * CC;
    float acc = b_out[f];
    #pragma unroll
    for (int c = 0; c < CC; ++c) acc += hp[c] * W_out[c * FF + f];
    out[tid] = acc;
}

// ---------------- host orchestration ----------------
extern "C" void kernel_launch(void* const* d_in, const int* in_sizes, int n_in,
                              void* d_out, int out_size, void* d_ws, size_t ws_size,
                              hipStream_t stream) {
    const float* x     = (const float*)d_in[0];
    const float* W_in  = (const float*)d_in[1];
    const float* b_in  = (const float*)d_in[2];
    const float* W_out = (const float*)d_in[3];
    const float* b_out = (const float*)d_in[4];
    const float* Wq    = (const float*)d_in[5];
    const float* bq    = (const float*)d_in[6];
    const float* Wk    = (const float*)d_in[7];
    const float* bk    = (const float*)d_in[8];
    const float* Wv    = (const float*)d_in[9];
    const float* bv    = (const float*)d_in[10];
    float* out = (float*)d_out;

    // workspace carve-out (256B aligned)
    char* base = (char*)d_ws;
    size_t off = 0;
    auto carve = [&](size_t bytes) -> char* {
        char* p = base + off;
        off = (off + bytes + 255) & ~(size_t)255;
        return p;
    };
    float*     stats = (float*)    carve(64);
    float*     hA    = (float*)    carve((size_t)PP * CC * 4);
    float*     hB    = (float*)    carve((size_t)PP * CC * 4);
    _Float16*  h16   = (_Float16*) carve((size_t)PP * CC * 2);
    float*     Qb    = (float*)    carve((size_t)PP * CC * 4);
    float*     Sb    = (float*)    carve((size_t)PP * HH * SSTRIDE * 4);
    _Float16*  kv16  = (_Float16*) carve((size_t)9 * PP * CC * 2);
    float*     Ob    = (float*)    carve((size_t)PP * CC * 4);
    _Float16*  w16   = (_Float16*) carve((size_t)NLAY * 55 * 4096 * 2);
    (void)ws_size; (void)in_sizes; (void)n_in; (void)out_size;

    bn_stats_k<<<1, 256, 0, stream>>>(x, stats);
    input_proj_k<<<(PP * CC) / 256, 256, 0, stream>>>(x, W_in, b_in, stats, hA, h16);
    wconv_k<<<(NLAY * 55 * 4096 + 255) / 256, 256, 0, stream>>>(Wq, Wk, Wv, w16);

    float* hcur = hA;
    float* hnext = hB;
    const int attnBlocks = (PP * HH) / 256;   // 512

    for (int l = 0; l < NLAY; ++l) {
        const _Float16* w16_l = w16 + (size_t)l * 55 * 4096;
        const float* bk_l = bk + (size_t)l * MM * CC;
        const float* bv_l = bv + (size_t)l * MM * CC;

        // Q projection (f32 output)
        gemm64_k<<<dim3(PP / 128, 1), 128, 0, stream>>>(
            h16, w16_l, bq + (size_t)l * CC, Qb, (_Float16*)nullptr, 0);

        // scores, time slice by time slice (K stays in L2)
        for (int t = 0; t < TK; ++t) {
            gemm64_k<<<dim3(PP / 128, 9), 128, 0, stream>>>(
                h16, w16_l + (size_t)(1 + t * 9) * 4096, bk_l + (size_t)t * 9 * CC,
                (float*)nullptr, kv16, 1);
            score_k<<<attnBlocks, 256, 0, stream>>>(Qb, kv16, bk_l + (size_t)t * 9 * CC, Sb, t);
        }

        softmax_k<<<attnBlocks, 256, 0, stream>>>(Sb);

        // weighted V accumulation, same chunking; last slice fuses residual
        for (int t = 0; t < TK; ++t) {
            gemm64_k<<<dim3(PP / 128, 9), 128, 0, stream>>>(
                h16, w16_l + (size_t)(28 + t * 9) * 4096, bv_l + (size_t)t * 9 * CC,
                (float*)nullptr, kv16, 1);
            accum_k<<<attnBlocks, 256, 0, stream>>>(
                Sb, kv16, bv_l + (size_t)t * 9 * CC, Ob,
                hcur, hnext, h16, t, t == 0 ? 1 : 0, t == TK - 1 ? 1 : 0);
        }
        { float* tmp = hcur; hcur = hnext; hnext = tmp; }
    }

    output_proj_k<<<(PP * FF) / 256, 256, 0, stream>>>(hcur, W_out, b_out, out);
}
